// Speaker_42116449304711
// MI455X (gfx1250) — compile-verified, matrix operand order checked by reading, not run on hardware
//
#include <hip/hip_runtime.h>

// ---------------------------------------------------------------------------
// Problem constants (from reference)
// ---------------------------------------------------------------------------
#define Bn      512
#define NOBJ    3
#define FIN     2048
#define FEAT    512
#define EMBED   512
#define HID     512
#define VOCAB   4096
#define MAXLEN  40
#define TSTEPS  38
#define SOS_IDX 1
#define EOS_IDX 2
#define KRAW    1539          // 3*(FEAT+1)
#define KFT     1568          // padded to 49*32 for WMMA K-steps

typedef __bf16 bf16;
typedef bf16  bf16x8 __attribute__((ext_vector_type(8)));
typedef bf16  v16bf  __attribute__((ext_vector_type(16)));
typedef float v8f    __attribute__((ext_vector_type(8)));
typedef int   v4i    __attribute__((vector_size(16)));

// ---------------------------------------------------------------------------
// CDNA5 async global->LDS copy (probe-verified signature on this toolchain:
// (v4i*, v4i*, imm offset, imm cpol), generic pointers). Falls back to a
// plain 16B copy if the builtin is absent.
// ---------------------------------------------------------------------------
#if defined(__AMDGCN__) && __has_builtin(__builtin_amdgcn_global_load_async_to_lds_b128)
#define USE_ASYNC_LDS 1
#else
#define USE_ASYNC_LDS 0
#endif

__device__ __forceinline__ void async_copy_b128(const bf16* g, bf16* l) {
#if USE_ASYNC_LDS
  __builtin_amdgcn_global_load_async_to_lds_b128((v4i*)g, (v4i*)l, 0, 0);
#else
  *(bf16x8*)l = *(const bf16x8*)g;
#endif
}

__device__ __forceinline__ void wait_async_lds() {
#if USE_ASYNC_LDS
#if __has_builtin(__builtin_amdgcn_s_wait_asynccnt)
  __builtin_amdgcn_s_wait_asynccnt(0);
#else
  asm volatile("s_wait_asynccnt 0" ::: "memory");
#endif
#endif
}

// ---------------------------------------------------------------------------
// bf16 WMMA GEMM:  C[M,N] = A[M,K] (row-major) @ W[N,K].T (row-major) + bias
// block = 256 threads (8 wave32); block tile 128(M) x 64(N); wave tile 16x64.
// grid = (N/64, M/128). K multiple of 32. Optionally mirrors C into bf16.
// Double-buffered LDS W-tile: async DMA of tile k+1 overlaps WMMA on tile k;
// exactly one barrier pair per K-step.
// ---------------------------------------------------------------------------
__global__ __launch_bounds__(256)
void gemm_bf16_wmma(const bf16* __restrict__ A, int lda,
                    const bf16* __restrict__ W, int ldw,
                    const float* __restrict__ bias,
                    float* __restrict__ C, bf16* __restrict__ Cbf, int ldc,
                    int K)
{
  __shared__ __align__(16) bf16 sB[2][64][40];  // 2 x (64 N-rows x 32 K, pad 40)

  const int lane = threadIdx.x & 31;
  const int wave = threadIdx.x >> 5;
  const int l15  = lane & 15;
  const int hl   = lane >> 4;                 // 0 or 1
  const int koff = hl * 8;                    // ISA 16-bit fragment K offset
  const int m0   = blockIdx.y * 128 + wave * 16;
  const int n0   = blockIdx.x * 64;

  v8f acc[4];
#pragma unroll
  for (int j = 0; j < 4; ++j)
#pragma unroll
    for (int e = 0; e < 8; ++e) acc[j][e] = 0.0f;

  const bf16* Arow = A + (size_t)(m0 + l15) * lda;
  const int srow = threadIdx.x >> 2;          // 0..63
  const int scol = (threadIdx.x & 3) * 8;     // 0,8,16,24
  const bf16* Wg  = W + (size_t)(n0 + srow) * ldw + scol;

  // prologue: stage first W tile into buffer 0
  async_copy_b128(Wg, &sB[0][srow][scol]);

  int buf = 0;
  for (int k = 0; k < K; k += 32) {
    wait_async_lds();          // our wave's in-flight DMA has landed
    __syncthreads();           // everyone's has; prior reads of other buf done

    const bf16 (*sBc)[40] = sB[buf];
    if (k + 32 < K) {          // kick off next tile into the other buffer;
      buf ^= 1;                // overlaps with the WMMAs below
      async_copy_b128(Wg + k + 32, &sB[buf][srow][scol]);
    }

    // A fragment: lane L<16 -> row m0+L, K = k..k+7, k+16..k+23
    //             lane L>=16 -> same row, K = k+8..k+15, k+24..k+31
    v16bf af;
    {
      bf16x8 a0 = *(const bf16x8*)(Arow + k + koff);
      bf16x8 a1 = *(const bf16x8*)(Arow + k + koff + 16);
#pragma unroll
      for (int i = 0; i < 8; ++i) { af[i] = a0[i]; af[8 + i] = a1[i]; }
    }

#pragma unroll
    for (int j = 0; j < 4; ++j) {
      const bf16* bp = &sBc[j * 16 + l15][koff];
      bf16x8 b0 = *(const bf16x8*)bp;
      bf16x8 b1 = *(const bf16x8*)(bp + 16);
      v16bf bfv;
#pragma unroll
      for (int i = 0; i < 8; ++i) { bfv[i] = b0[i]; bfv[8 + i] = b1[i]; }
      acc[j] = __builtin_amdgcn_wmma_f32_16x16x32_bf16(
          false, af, false, bfv, (short)0, acc[j], false, false);
    }
  }

  // D layout: VGPR g -> M = m0 + g + 8*hl, N = n0 + j*16 + l15
#pragma unroll
  for (int j = 0; j < 4; ++j) {
    const int n = n0 + j * 16 + l15;
    const float bv = bias ? bias[n] : 0.0f;
#pragma unroll
    for (int g = 0; g < 8; ++g) {
      const int m = m0 + g + hl * 8;
      const float v = acc[j][g] + bv;
      C[(size_t)m * ldc + n] = v;
      if (Cbf) Cbf[(size_t)m * ldc + n] = (bf16)v;
    }
  }
}

// ---------------------------------------------------------------------------
// Elementwise converts / packing
// ---------------------------------------------------------------------------
__global__ void cvt_bf16_k(const float* __restrict__ s, bf16* __restrict__ d, int n) {
  int i = blockIdx.x * 256 + threadIdx.x;
  if (i < n) d[i] = (bf16)s[i];
}

__global__ void cvt_initW_k(const float* __restrict__ s, bf16* __restrict__ d) {
  int i = blockIdx.x * 256 + threadIdx.x;
  if (i < HID * KFT) {
    int r = i / KFT, c = i - r * KFT;
    d[i] = (c < KRAW) ? (bf16)s[(size_t)r * KRAW + c] : (bf16)0.0f;
  }
}

// ft[b] = [fe[b,0,:], 1{tgt==0}, fe[b,1,:], 1{tgt==1}, fe[b,2,:], 1{tgt==2}, 0-pad]
__global__ void pack_ft_k(const float* __restrict__ fe, const int* __restrict__ tgt,
                          bf16* __restrict__ ft) {
  int i = blockIdx.x * 256 + threadIdx.x;
  if (i < Bn * KFT) {
    int b = i / KFT, c = i - b * KFT;
    float v = 0.0f;
    if (c < KRAW) {
      int o = c / (FEAT + 1), j = c - o * (FEAT + 1);
      v = (j < FEAT) ? fe[((size_t)b * NOBJ + o) * FEAT + j]
                     : (tgt[b] == o ? 1.0f : 0.0f);
    }
    ft[i] = (bf16)v;
  }
}

// ---------------------------------------------------------------------------
// GRU gate pointwise:  h <- (1-z)*n + z*h ; also mirror to bf16
// ---------------------------------------------------------------------------
__global__ void gru_gate_k(const float* __restrict__ gi, const float* __restrict__ gh,
                           float* __restrict__ h, bf16* __restrict__ hb) {
  int idx = blockIdx.x * 256 + threadIdx.x;
  if (idx >= Bn * HID) return;
  int b = idx / HID, j = idx - b * HID;
  size_t base = (size_t)b * 3 * HID;
  float ir = gi[base + j], iz = gi[base + HID + j], in_ = gi[base + 2 * HID + j];
  float hr = gh[base + j], hz = gh[base + HID + j], hn  = gh[base + 2 * HID + j];
  float r  = 1.0f / (1.0f + __expf(-(ir + hr)));
  float z  = 1.0f / (1.0f + __expf(-(iz + hz)));
  float nn = tanhf(in_ + r * hn);
  float hv = (1.0f - z) * nn + z * h[idx];
  h[idx]  = hv;
  hb[idx] = (bf16)hv;
}

// ---------------------------------------------------------------------------
// Init: x0 = emb_W[SOS]; lang[b,0,SOS]=1; done=0; len=1
// ---------------------------------------------------------------------------
__global__ void init_state_k(const float* __restrict__ embW, bf16* __restrict__ xb,
                             float* __restrict__ lang, int* __restrict__ done,
                             int* __restrict__ len) {
  int idx = blockIdx.x * 256 + threadIdx.x;
  if (idx < Bn * EMBED) {
    int b = idx / EMBED, e = idx - b * EMBED;
    xb[idx] = (bf16)embW[(size_t)SOS_IDX * EMBED + e];
    if (e == 0) {
      lang[(size_t)b * MAXLEN * VOCAB + SOS_IDX] = 1.0f;
      done[b] = 0;
      len[b]  = 1;
    }
  }
}

// ---------------------------------------------------------------------------
// Per-step sample: online softmax+argmax over (logits+gumbel), straight-through
// scale c=(1+s)-s, write lang[b,t+1,pred]=c (if not done), x <- c*emb_W[pred].
// One block (256 threads, 8 waves) per batch row.
// ---------------------------------------------------------------------------
__global__ __launch_bounds__(256)
void sample_step_k(const float* __restrict__ logits, const float* __restrict__ gum,
                   const bf16* __restrict__ embWb, float* __restrict__ lang,
                   int* __restrict__ done, int* __restrict__ len,
                   bf16* __restrict__ xb, int t)
{
  const int b = blockIdx.x;
  const float* lrow = logits + (size_t)b * VOCAB;
  const float* grow = gum    + (size_t)b * VOCAB;

  float m = -3.4e38f, s = 0.0f;
  int arg = 0x7fffffff;
  for (int v = threadIdx.x; v < VOCAB; v += 256) {
    float z = lrow[v] + grow[v];
    if (z > m) { s = s * __expf(m - z) + 1.0f; m = z; arg = v; }
    else       { s += __expf(z - m); }
  }
  // wave32 reduction
  for (int off = 16; off; off >>= 1) {
    float om = __shfl_xor(m, off);
    float os = __shfl_xor(s, off);
    int   oa = __shfl_xor(arg, off);
    if (om > m || (om == m && oa < arg)) { s = os + s * __expf(m - om); m = om; arg = oa; }
    else                                 { s = s + os * __expf(om - m); }
  }
  __shared__ float wm[8], wsum[8];
  __shared__ int   warg[8];
  if ((threadIdx.x & 31) == 0) {
    wm[threadIdx.x >> 5] = m; wsum[threadIdx.x >> 5] = s; warg[threadIdx.x >> 5] = arg;
  }
  __syncthreads();
  float M = wm[0], S = wsum[0];
  int   A0 = warg[0];
  for (int w = 1; w < 8; ++w) {
    float om = wm[w], os = wsum[w];
    int   oa = warg[w];
    if (om > M || (om == M && oa < A0)) { S = os + S * __expf(M - om); M = om; A0 = oa; }
    else                                { S = S + os * __expf(om - M); }
  }

  const float sm = 1.0f / S;           // softmax value at argmax (TAU=1)
  const float c  = (1.0f + sm) - sm;   // straight-through scale (exact fp semantics)

  if (threadIdx.x == 0) {
    if (!done[b]) {
      len[b] += 1;
      lang[((size_t)b * MAXLEN + (t + 1)) * VOCAB + A0] = c;
      if (A0 == EOS_IDX) done[b] = 1;
    }
  }
  const bf16* er = embWb + (size_t)A0 * EMBED;
  for (int e = threadIdx.x; e < EMBED; e += 256)
    xb[(size_t)b * EMBED + e] = (bf16)(c * (float)er[e]);
}

// ---------------------------------------------------------------------------
// Finalize: appended EOS + length output + eos_loss
// ---------------------------------------------------------------------------
__global__ void finalize_k(float* __restrict__ lang, const int* __restrict__ done,
                           const int* __restrict__ len, float* __restrict__ out_len,
                           float* __restrict__ out_eos) {
  int b = blockIdx.x * 256 + threadIdx.x;
  if (b < Bn) {
    int L = len[b] + (done[b] ? 0 : 1);
    if (!done[b])
      lang[((size_t)b * MAXLEN + (MAXLEN - 1)) * VOCAB + EOS_IDX] = 1.0f;
    out_len[b] = (float)L;
    if (b == 0) out_eos[0] = 0.0f;
  }
}

// ---------------------------------------------------------------------------
// Host orchestration
// ---------------------------------------------------------------------------
extern "C" void kernel_launch(void* const* d_in, const int* in_sizes, int n_in,
                              void* d_out, int out_size, void* d_ws, size_t ws_size,
                              hipStream_t stream) {
  (void)in_sizes; (void)n_in; (void)out_size; (void)ws_size;

  const float* feats   = (const float*)d_in[0];
  const int*   targets = (const int*)  d_in[1];
  const float* feat_W  = (const float*)d_in[2];
  const float* feat_b  = (const float*)d_in[3];
  const float* emb_W   = (const float*)d_in[4];
  const float* init_W  = (const float*)d_in[5];
  const float* init_b  = (const float*)d_in[6];
  const float* gru_Wih = (const float*)d_in[7];
  const float* gru_Whh = (const float*)d_in[8];
  const float* gru_bih = (const float*)d_in[9];
  const float* gru_bhh = (const float*)d_in[10];
  const float* out_W   = (const float*)d_in[11];
  const float* out_b   = (const float*)d_in[12];
  const float* gumbel  = (const float*)d_in[13];

  // workspace carve (256B aligned)
  char* wp = (char*)d_ws;
  auto carve = [&](size_t bytes) -> char* {
    char* p = wp; wp += (bytes + 255) & ~(size_t)255; return p;
  };
  bf16* featsb = (bf16*)carve((size_t)Bn * NOBJ * FIN * 2);
  bf16* featWb = (bf16*)carve((size_t)FEAT * FIN * 2);
  bf16* initWb = (bf16*)carve((size_t)HID * KFT * 2);
  bf16* Wihb   = (bf16*)carve((size_t)3 * HID * EMBED * 2);
  bf16* Whhb   = (bf16*)carve((size_t)3 * HID * HID * 2);
  bf16* outWb  = (bf16*)carve((size_t)VOCAB * HID * 2);
  bf16* embWb  = (bf16*)carve((size_t)VOCAB * EMBED * 2);
  bf16* ftb    = (bf16*)carve((size_t)Bn * KFT * 2);
  float* fe    = (float*)carve((size_t)Bn * NOBJ * FEAT * 4);
  float* h     = (float*)carve((size_t)Bn * HID * 4);
  bf16*  hb    = (bf16*) carve((size_t)Bn * HID * 2);
  bf16*  xb    = (bf16*) carve((size_t)Bn * EMBED * 2);
  float* gi    = (float*)carve((size_t)Bn * 3 * HID * 4);
  float* gh    = (float*)carve((size_t)Bn * 3 * HID * 4);
  float* lgts  = (float*)carve((size_t)Bn * VOCAB * 4);
  int*   done  = (int*)  carve((size_t)Bn * 4);
  int*   len   = (int*)  carve((size_t)Bn * 4);

  float* lang    = (float*)d_out;                        // [B, MAXLEN, VOCAB]
  float* out_len = lang + (size_t)Bn * MAXLEN * VOCAB;   // [B]
  float* out_eos = out_len + Bn;                         // [1]

  (void)hipMemsetAsync(lang, 0, (size_t)Bn * MAXLEN * VOCAB * sizeof(float), stream);

  auto cvt = [&](const float* s, bf16* d, int n) {
    cvt_bf16_k<<<(n + 255) / 256, 256, 0, stream>>>(s, d, n);
  };
  cvt(feats,   featsb, Bn * NOBJ * FIN);
  cvt(feat_W,  featWb, FEAT * FIN);
  cvt(gru_Wih, Wihb,   3 * HID * EMBED);
  cvt(gru_Whh, Whhb,   3 * HID * HID);
  cvt(out_W,   outWb,  VOCAB * HID);
  cvt(emb_W,   embWb,  VOCAB * EMBED);
  cvt_initW_k<<<(HID * KFT + 255) / 256, 256, 0, stream>>>(init_W, initWb);

  // fe = feats @ feat_W.T + feat_b        [1536,2048] x [512,2048]^T
  gemm_bf16_wmma<<<dim3(FEAT / 64, (Bn * NOBJ) / 128), 256, 0, stream>>>(
      featsb, FIN, featWb, FIN, feat_b, fe, nullptr, FEAT, FIN);
  pack_ft_k<<<(Bn * KFT + 255) / 256, 256, 0, stream>>>(fe, targets, ftb);

  // h0 = ft @ init_W.T + init_b           [512,1568] x [512,1568]^T
  gemm_bf16_wmma<<<dim3(HID / 64, Bn / 128), 256, 0, stream>>>(
      ftb, KFT, initWb, KFT, init_b, h, hb, HID, KFT);

  init_state_k<<<(Bn * EMBED + 255) / 256, 256, 0, stream>>>(emb_W, xb, lang, done, len);

  for (int t = 0; t < TSTEPS; ++t) {
    // gi = x @ Wih.T + bih               [512,512] x [1536,512]^T
    gemm_bf16_wmma<<<dim3((3 * HID) / 64, Bn / 128), 256, 0, stream>>>(
        xb, EMBED, Wihb, EMBED, gru_bih, gi, nullptr, 3 * HID, EMBED);
    // gh = h @ Whh.T + bhh
    gemm_bf16_wmma<<<dim3((3 * HID) / 64, Bn / 128), 256, 0, stream>>>(
        hb, HID, Whhb, HID, gru_bhh, gh, nullptr, 3 * HID, HID);
    gru_gate_k<<<(Bn * HID + 255) / 256, 256, 0, stream>>>(gi, gh, h, hb);
    // logits = h @ out_W.T + out_b       [512,512] x [4096,512]^T
    gemm_bf16_wmma<<<dim3(VOCAB / 64, Bn / 128), 256, 0, stream>>>(
        hb, HID, outWb, HID, out_b, lgts, nullptr, VOCAB, HID);
    sample_step_k<<<Bn, 256, 0, stream>>>(
        lgts, gumbel + (size_t)t * Bn * VOCAB, embWb, lang, done, len, xb, t);
  }

  finalize_k<<<(Bn + 255) / 256, 256, 0, stream>>>(lang, done, len, out_len, out_eos);
}
